// KarateGraph3SAGE_68599217652368
// MI455X (gfx1250) — compile-verified, matrix operand order checked by reading, not run on hardware
//
#include <hip/hip_runtime.h>

typedef __attribute__((ext_vector_type(16))) __bf16 v16bf;
typedef __attribute__((ext_vector_type(8)))  __bf16 v8bf;   // 16 bytes -> one b128
typedef __attribute__((ext_vector_type(4)))  __bf16 v4bf;   // 8 bytes
typedef __attribute__((ext_vector_type(8)))  float  v8f;

#define N_NODES 10000

union Frag { v16bf v; v8bf h[2]; };

// ---------------------------------------------------------------- utilities

__global__ void zero_f32(float* __restrict__ p, int n) {
  int i = blockIdx.x * blockDim.x + threadIdx.x;
  if (i < n) p[i] = 0.0f;
}

__global__ void degree_count(const int* __restrict__ dst, float* __restrict__ cnt, int E) {
  int e = blockIdx.x * blockDim.x + threadIdx.x;
  if (e < E) atomicAdd(&cnt[dst[e]], 1.0f);
}

// one thread per (edge, 4-float chunk): coalesced gather from src row,
// 4 global_atomic_add_f32 into dst row (L2-side RMW atomics)
__global__ void scatter_add(const float* __restrict__ x, const int* __restrict__ src,
                            const int* __restrict__ dst, float* __restrict__ agg,
                            int E, int F4) {
  long long t = (long long)blockIdx.x * blockDim.x + threadIdx.x;
  long long total = (long long)E * F4;
  if (t >= total) return;
  int e = (int)(t / F4);
  int c = (int)(t % F4);
  int s = src[e], d = dst[e];
  const float4 v = ((const float4*)(x + (long long)s * F4 * 4))[c];
  float* o = agg + (long long)d * F4 * 4 + (long long)c * 4;
  atomicAdd(o + 0, v.x);
  atomicAdd(o + 1, v.y);
  atomicAdd(o + 2, v.z);
  atomicAdd(o + 3, v.w);
}

// agg[row][f] /= max(cnt[row], 1)
__global__ void finalize_mean(float* __restrict__ agg, const float* __restrict__ cnt,
                              int F, int total) {
  int i = blockIdx.x * blockDim.x + threadIdx.x;
  if (i >= total) return;
  int row = i / F;
  agg[i] *= (1.0f / fmaxf(cnt[row], 1.0f));
}

// f32 -> bf16 row-major copy (4 elements / thread; n multiple of 4)
__global__ void cvt_bf16(const float* __restrict__ in, __bf16* __restrict__ out, int n) {
  int i4 = (blockIdx.x * blockDim.x + threadIdx.x) * 4;
  if (i4 >= n) return;
  float4 v = *(const float4*)(in + i4);
  v4bf o = { (__bf16)v.x, (__bf16)v.y, (__bf16)v.z, (__bf16)v.w };
  *(v4bf*)(out + i4) = o;
}

// Pack W[K,O] (f32 row-major) into fragment-major bf16:
// P[((kt*(O/16)+nt)*32 + lane)*16 + j] = lane's j-th B-fragment element
// (ISA 7.12.2 16-bit B layout: lane -> col n=lane&15, half=lane>>4 selects
//  K-runs [kt*32+half*8,+8) and [kt*32+16+half*8,+8))
__global__ void pack_w_bf16(const float* __restrict__ W, __bf16* __restrict__ P,
                            int K, int O) {
  int t = blockIdx.x * blockDim.x + threadIdx.x;
  int ntilesO = O >> 4;
  int total = (K >> 5) * ntilesO * 32;
  if (t >= total) return;
  int lane = t & 31;
  int frag = t >> 5;
  int nt = frag % ntilesO;
  int kt = frag / ntilesO;
  int n  = nt * 16 + (lane & 15);
  int kb = kt * 32 + (lane >> 4) * 8;
  __bf16* o = P + ((size_t)t << 4);
#pragma unroll
  for (int j = 0; j < 8; ++j) {
    o[j]     = (__bf16)W[(size_t)(kb + j) * O + n];
    o[j + 8] = (__bf16)W[(size_t)(kb + 16 + j) * O + n];
  }
}

// ---------------------------------------------------------------- WMMA GEMM
// out[M,O] = relu?( Am@Wl + Sm@Wr + bias ), M = N_NODES (625 tiles of 16).
// Each wave owns a 16x64 strip. All pointers are strength-reduced running
// pointers; each k-step hoists 20 b128 loads, then issues 8 WMMAs.
template <int RELU>
__launch_bounds__(256)
__global__ void sage_gemm_wmma(const __bf16* __restrict__ Am, const __bf16* __restrict__ Sm,
                               const __bf16* __restrict__ WlP, const __bf16* __restrict__ WrP,
                               const float* __restrict__ bias, float* __restrict__ out,
                               int K, int O) {
  const int NT = 4;
  int wave = (blockIdx.x * blockDim.x + threadIdx.x) >> 5;
  int lane = threadIdx.x & 31;
  int tilesN = O >> 6;            // strips of 64 columns
  int tilesM = N_NODES >> 4;      // 625
  int tm = wave / tilesN;
  int tn = wave % tilesN;
  if (tm >= tilesM) return;

  int l15 = lane & 15;
  int hh  = lane >> 4;

  // running pointers (element units); per k-tile increments are constants
  const __bf16* amp = Am + (size_t)(tm * 16 + l15) * K + hh * 8;   // += 32
  const __bf16* asp = Sm + (size_t)(tm * 16 + l15) * K + hh * 8;   // += 32
  const size_t wkStride = (size_t)(O >> 4) << 9;                   // ntilesO*32*16
  const __bf16* wlp = WlP + (((size_t)tn * 4 * 32 + lane) << 4);   // += wkStride
  const __bf16* wrp = WrP + (((size_t)tn * 4 * 32 + lane) << 4);   // += wkStride

  v8f zero = {};
  v8f acc[NT];
#pragma unroll
  for (int t = 0; t < NT; ++t) acc[t] = zero;

  int ktiles = K >> 5;
  for (int kt = 0; kt < ktiles; ++kt) {
    // ---- hoisted loads: 4x b128 (A) + 16x b128 (B), all immediate offsets
    Frag aM, aS, bl[NT], br[NT];
    aM.h[0] = *(const v8bf*)(amp);
    aM.h[1] = *(const v8bf*)(amp + 16);
    aS.h[0] = *(const v8bf*)(asp);
    aS.h[1] = *(const v8bf*)(asp + 16);
#pragma unroll
    for (int t = 0; t < NT; ++t) {
      bl[t].h[0] = *(const v8bf*)(wlp + t * 512);
      bl[t].h[1] = *(const v8bf*)(wlp + t * 512 + 8);
      br[t].h[0] = *(const v8bf*)(wrp + t * 512);
      br[t].h[1] = *(const v8bf*)(wrp + t * 512 + 8);
    }
    amp += 32;
    asp += 32;
    wlp += wkStride;
    wrp += wkStride;
    // ---- 8 back-to-back WMMAs on the loaded fragments
#pragma unroll
    for (int t = 0; t < NT; ++t)
      acc[t] = __builtin_amdgcn_wmma_f32_16x16x32_bf16(
          false, aM.v, false, bl[t].v, (short)0, acc[t], false, false);
#pragma unroll
    for (int t = 0; t < NT; ++t)
      acc[t] = __builtin_amdgcn_wmma_f32_16x16x32_bf16(
          false, aS.v, false, br[t].v, (short)0, acc[t], false, false);
  }

  // C/D layout: VGPR r -> row tm*16 + hh*8 + r, col = tn*64 + t*16 + l15
  int row0 = tm * 16 + hh * 8;
#pragma unroll
  for (int t = 0; t < NT; ++t) {
    int n = tn * 64 + t * 16 + l15;
    float bv = bias[n];
#pragma unroll
    for (int r = 0; r < 8; ++r) {
      float v = acc[t][r] + bv;
      if (RELU) v = fmaxf(v, 0.0f);
      out[(size_t)(row0 + r) * O + n] = v;
    }
  }
}

// ------------------------------------------------------------- log_softmax
// 64 columns per row: one wave32 per row, 2 elements per lane.
__global__ void logsoftmax64(const float* __restrict__ in, float* __restrict__ out, int M) {
  int row = blockIdx.x * (blockDim.x >> 5) + (threadIdx.x >> 5);
  if (row >= M) return;
  int lane = threadIdx.x & 31;
  float a = in[(size_t)row * 64 + lane];
  float b = in[(size_t)row * 64 + lane + 32];
  float mx = fmaxf(a, b);
#pragma unroll
  for (int o = 16; o > 0; o >>= 1) mx = fmaxf(mx, __shfl_xor(mx, o, 32));
  float s = expf(a - mx) + expf(b - mx);
#pragma unroll
  for (int o = 16; o > 0; o >>= 1) s += __shfl_xor(s, o, 32);
  float lse = logf(s) + mx;
  out[(size_t)row * 64 + lane]      = a - lse;
  out[(size_t)row * 64 + lane + 32] = b - lse;
}

// ---------------------------------------------------------------- launcher

static inline int cdiv_ll(long long a, long long b) { return (int)((a + b - 1) / b); }

extern "C" void kernel_launch(void* const* d_in, const int* in_sizes, int n_in,
                              void* d_out, int out_size, void* d_ws, size_t ws_size,
                              hipStream_t stream) {
  const float* x   = (const float*)d_in[0];
  const int*   ei  = (const int*)d_in[1];   // [2,E] int32
  const float* w1l = (const float*)d_in[2];
  const float* w1r = (const float*)d_in[3];
  const float* b1  = (const float*)d_in[4];
  const float* w2l = (const float*)d_in[5];
  const float* w2r = (const float*)d_in[6];
  const float* b2  = (const float*)d_in[7];
  const float* w3l = (const float*)d_in[8];
  const float* w3r = (const float*)d_in[9];
  const float* b3  = (const float*)d_in[10];
  float* out = (float*)d_out;

  const int E = in_sizes[1] / 2;
  const int* src = ei;
  const int* dst = ei + E;

  // workspace layout: f32 regions, then bf16 regions (~134 MB total)
  float* cnt = (float*)d_ws;                         // 10240 f32
  float* agg = cnt + 10240;                          // 10000*1024 f32
  float* h1  = agg + (size_t)N_NODES * 1024;         // 10000*256  f32
  float* h2  = h1  + (size_t)N_NODES * 256;          // 10000*1024 f32
  __bf16* aBf = (__bf16*)(h2 + (size_t)N_NODES * 1024); // 10000*1024 bf16 (mean)
  __bf16* sBf = aBf + (size_t)N_NODES * 1024;           // 10000*1024 bf16 (self)
  __bf16* wlP = sBf + (size_t)N_NODES * 1024;           // 256*1024 bf16 packed
  __bf16* wrP = wlP + (size_t)256 * 1024;               // 256*1024 bf16 packed

  // degree counts (shared by all three layers)
  zero_f32<<<cdiv_ll(N_NODES, 256), 256, 0, stream>>>(cnt, N_NODES);
  degree_count<<<cdiv_ll(E, 256), 256, 0, stream>>>(dst, cnt, E);

  struct Layer {
    const float* self; const float* wl; const float* wr; const float* bias;
    float* o; int F; int O; int relu;
  };
  const Layer layers[3] = {
    { x,  w1l, w1r, b1, h1, 128,  256,  1 },
    { h1, w2l, w2r, b2, h2, 256,  1024, 1 },
    { h2, w3l, w3r, b3, h1, 1024, 64,   0 },  // logits reuse h1
  };

  for (int li = 0; li < 3; ++li) {
    const Layer& L = layers[li];
    const int NF = N_NODES * L.F;
    // mean aggregation into agg
    zero_f32<<<cdiv_ll(NF, 256), 256, 0, stream>>>(agg, NF);
    scatter_add<<<cdiv_ll((long long)E * (L.F / 4), 256), 256, 0, stream>>>(
        L.self, src, dst, agg, E, L.F / 4);
    finalize_mean<<<cdiv_ll(NF, 256), 256, 0, stream>>>(agg, cnt, L.F, NF);
    // bf16 staging: activations row-major, weights fragment-major
    cvt_bf16<<<cdiv_ll(NF / 4, 256), 256, 0, stream>>>(agg, aBf, NF);
    cvt_bf16<<<cdiv_ll(NF / 4, 256), 256, 0, stream>>>(L.self, sBf, NF);
    const int wfrag = (L.F / 32) * (L.O / 16) * 32;
    pack_w_bf16<<<cdiv_ll(wfrag, 256), 256, 0, stream>>>(L.wl, wlP, L.F, L.O);
    pack_w_bf16<<<cdiv_ll(wfrag, 256), 256, 0, stream>>>(L.wr, wrP, L.F, L.O);
    // fused dual GEMM (+bias, +relu)
    long long waves = (long long)(N_NODES / 16) * (L.O / 64);
    if (L.relu)
      sage_gemm_wmma<1><<<cdiv_ll(waves * 32, 256), 256, 0, stream>>>(
          aBf, sBf, wlP, wrP, L.bias, L.o, L.F, L.O);
    else
      sage_gemm_wmma<0><<<cdiv_ll(waves * 32, 256), 256, 0, stream>>>(
          aBf, sBf, wlP, wrP, L.bias, L.o, L.F, L.O);
  }

  // log_softmax over 64 logits per row (logits are in h1)
  logsoftmax64<<<cdiv_ll((long long)N_NODES * 32, 256), 256, 0, stream>>>(h1, out, N_NODES);
}
